// DistanceAwareMultiheadAttention_53403623358963
// MI455X (gfx1250) — matrix operand order repaired and checked by reading, not compiled, not run on hardware
//
#include <hip/hip_runtime.h>

// ---------------------------------------------------------------------------
// DistanceAwareMultiheadAttention for MI455X (gfx1250, wave32, WMMA)
//   B=4, S=1024, D=768, H=12, DK=64, NUM_EMB=10
// Pipeline:
//   1) proj_kernel: Q/K/V = X*W^T + b  (bf16 WMMA, fp32 acc)
//        Q,K stored bf16 [B,H,S,DK];  V stored bf16 TRANSPOSED [B,H,DK,S]
//   2) bias_table_kernel: T[b,h,s,e] = q . emb_e   ([B,H,S,10] fp32)
//   3) idx_kernel: idx[b,q,k] = clip(round(9*dist/MAXD),0,9)  (u8)
//   4) attn_kernel: flash-style attention; K/V tiles streamed into LDS with
//        GLOBAL_LOAD_ASYNC_TO_LDS_B128 (ASYNCcnt) and double-buffered;
//        all matmuls via v_wmma_f32_16x16x32_bf16; online softmax.
// ---------------------------------------------------------------------------

#define B_  4
#define S_  1024
#define D_  768
#define H_  12
#define DK_ 64
#define NE_ 10

typedef __bf16 bf16_t;
typedef __attribute__((ext_vector_type(16))) __bf16 v16bf;
typedef __attribute__((ext_vector_type(8)))  __bf16 v8bf;
typedef __attribute__((ext_vector_type(4)))  __bf16 v4bf;
typedef __attribute__((ext_vector_type(8)))  float  v8f;

static __device__ inline v16bf make_frag(const bf16_t* lo, const bf16_t* hi) {
  v8bf a = *(const v8bf*)lo;
  v8bf b = *(const v8bf*)hi;
  v16bf r;
#pragma unroll
  for (int i = 0; i < 8; ++i) { r[i] = a[i]; r[i + 8] = b[i]; }
  return r;
}

static __device__ inline v8f wmma_bf16(v16bf a, v16bf b, v8f c) {
  return __builtin_amdgcn_wmma_f32_16x16x32_bf16(
      /*neg_a=*/false, a, /*neg_b=*/false, b,
      /*c_mod=*/(short)0, c, /*reuse_a=*/false, /*reuse_b=*/false);
}

// Async 16B global->LDS copy (GVS mode: saddr base + 32-bit vgpr offset).
// vdst = LDS byte address (low 32 bits of generic pointer = LDS offset).
static __device__ inline void async_copy_b128(void* lds_dst,
                                              const void* gbase,
                                              unsigned byte_off) {
  unsigned ldsa = (unsigned)(unsigned long long)lds_dst;
  asm volatile("global_load_async_to_lds_b128 %0, %1, %2"
               :
               : "v"(ldsa), "v"(byte_off), "s"(gbase)
               : "memory");
}

static __device__ inline void wait_asynccnt0() {
  asm volatile("s_wait_asynccnt 0x0" ::: "memory");
}

// ---------------------------------------------------------------------------
// 1) Fused projection GEMM:  Y[n,o] = sum_d X[n,d]*W[o,d] + b[o]
//    block = 64 rows x 64 cols, 4 waves (wave w owns rows w*16..w*16+15)
//    z: 0=Q, 1=K, 2=V(transposed output layout)
// ---------------------------------------------------------------------------
__global__ __launch_bounds__(128) void proj_kernel(
    const float* __restrict__ Xq, const float* __restrict__ Xk,
    const float* __restrict__ Xv,
    const float* __restrict__ Wq, const float* __restrict__ Wk,
    const float* __restrict__ Wv,
    const float* __restrict__ bq, const float* __restrict__ bk,
    const float* __restrict__ bv,
    bf16_t* __restrict__ Qo, bf16_t* __restrict__ Ko,
    bf16_t* __restrict__ Vo) {
  const int z = blockIdx.z;
  const float* X  = (z == 0) ? Xq : (z == 1) ? Xk : Xv;
  const float* W  = (z == 0) ? Wq : (z == 1) ? Wk : Wv;
  const float* bb = (z == 0) ? bq : (z == 1) ? bk : bv;
  bf16_t* dst = (z == 0) ? Qo : (z == 1) ? Ko : Vo;
  const bool vtrans = (z == 2);

  __shared__ bf16_t xbuf[64 * 40];   // [row][d] pad 8
  __shared__ bf16_t wbuf[64 * 40];   // [o][d]  pad 8

  const int tid  = threadIdx.x;
  const int lane = tid & 31;
  const int wv   = tid >> 5;
  const int half = lane >> 4;
  const int n    = lane & 15;

  const int row0 = blockIdx.x * 64;
  const int col0 = blockIdx.y * 64;

  v8f acc[4];
#pragma unroll
  for (int g = 0; g < 4; ++g)
#pragma unroll
    for (int r = 0; r < 8; ++r) acc[g][r] = 0.0f;

  for (int kk = 0; kk < D_ / 32; ++kk) {
    // stage X chunk [64 rows x 32 d] and W chunk [64 o x 32 d] as bf16
#pragma unroll
    for (int i = 0; i < 4; ++i) {
      int c = i * 128 + tid;              // 0..511
      int r = c >> 3;
      int cc = (c & 7) * 4;
      float4 xv = *(const float4*)(X + (size_t)(row0 + r) * D_ + kk * 32 + cc);
      v4bf xb; xb[0] = (bf16_t)xv.x; xb[1] = (bf16_t)xv.y;
               xb[2] = (bf16_t)xv.z; xb[3] = (bf16_t)xv.w;
      *(v4bf*)&xbuf[r * 40 + cc] = xb;
      float4 wvv = *(const float4*)(W + (size_t)(col0 + r) * D_ + kk * 32 + cc);
      v4bf wb2; wb2[0] = (bf16_t)wvv.x; wb2[1] = (bf16_t)wvv.y;
                wb2[2] = (bf16_t)wvv.z; wb2[3] = (bf16_t)wvv.w;
      *(v4bf*)&wbuf[r * 40 + cc] = wb2;
      // prefetch next chunk (global_prefetch_b8)
      if (kk + 1 < D_ / 32) {
        __builtin_prefetch(X + (size_t)(row0 + r) * D_ + (kk + 1) * 32 + cc, 0, 1);
        __builtin_prefetch(W + (size_t)(col0 + r) * D_ + (kk + 1) * 32 + cc, 0, 1);
      }
    }
    __syncthreads();

    // A fragment: rows wv*16 + m
    const bf16_t* ab = &xbuf[(wv * 16 + n) * 40];
    v16bf af = make_frag(ab + half * 16, ab + half * 16 + 8);
#pragma unroll
    for (int g = 0; g < 4; ++g) {
      const bf16_t* wb = &wbuf[(g * 16 + n) * 40 + half * 16];
      v16bf bf_ = make_frag(wb, wb + 8);
      acc[g] = wmma_bf16(af, bf_, acc[g]);
    }
    __syncthreads();
  }

  const int bIdx  = row0 / S_;
  const int sBase = (row0 % S_) + wv * 16;
#pragma unroll
  for (int g = 0; g < 4; ++g) {
    const int o  = col0 + g * 16 + n;
    const float bo = bb[o];
    const int h  = o >> 6;
    const int dc = o & 63;
    if (!vtrans) {
      // [B,H,S,DK]
#pragma unroll
      for (int r = 0; r < 8; ++r) {
        int s = sBase + r + 8 * half;
        dst[(((size_t)(bIdx * H_ + h)) * S_ + s) * DK_ + dc] =
            (bf16_t)(acc[g][r] + bo);
      }
    } else {
      // [B,H,DK,S] : 8 consecutive s per half -> one 16B store
      v8bf pk;
#pragma unroll
      for (int r = 0; r < 8; ++r) pk[r] = (bf16_t)(acc[g][r] + bo);
      size_t off = (((size_t)(bIdx * H_ + h)) * DK_ + dc) * S_ +
                   sBase + 8 * half;
      *(v8bf*)(dst + off) = pk;
    }
  }
}

// ---------------------------------------------------------------------------
// 2) T[b,h,s,e] = q[b,h,s,:] . emb_k[e,:]
// ---------------------------------------------------------------------------
__global__ void bias_table_kernel(const bf16_t* __restrict__ Q,
                                  const float* __restrict__ emb,
                                  float* __restrict__ T) {
  int i = blockIdx.x * blockDim.x + threadIdx.x;
  if (i >= B_ * H_ * S_ * NE_) return;
  int e = i % NE_;
  int row = i / NE_;
  const bf16_t* qp = Q + (size_t)row * DK_;
  const float* ep = emb + e * DK_;
  float s = 0.0f;
#pragma unroll 8
  for (int d = 0; d < DK_; ++d) s += (float)qp[d] * ep[d];
  T[i] = s;
}

// ---------------------------------------------------------------------------
// 3) idx[b,q,k] from tile positions
// ---------------------------------------------------------------------------
__global__ void idx_kernel(const float* __restrict__ pos,
                           unsigned char* __restrict__ idx, long total) {
  long i = (long)blockIdx.x * blockDim.x + threadIdx.x;
  if (i >= total) return;
  int k = (int)(i % S_);
  long t = i / S_;
  int q = (int)(t % S_);
  int b = (int)(t / S_);
  const float* qp = pos + ((size_t)b * S_ + q) * 2;
  const float* kp = pos + ((size_t)b * S_ + k) * 2;
  float dx = qp[0] - kp[0], dy = qp[1] - kp[1];
  const float maxd = 100000.0f * 1.41421356237f;
  float d = sqrtf(dx * dx + dy * dy) * ((float)(NE_ - 1) / maxd);
  float r = roundf(d);
  r = fminf(fmaxf(r, 0.0f), (float)(NE_ - 1));
  idx[i] = (unsigned char)r;
}

// ---------------------------------------------------------------------------
// 4) Attention. grid = (S/64, H, B), block = 128 (4 waves).
//    Wave w processes q rows q0 = blockIdx.x*64 + w*16. Waves share K/V
//    tiles, double-buffered in LDS and filled by async global->LDS copies.
// ---------------------------------------------------------------------------
__global__ __launch_bounds__(128) void attn_kernel(
    const bf16_t* __restrict__ Q, const bf16_t* __restrict__ K,
    const bf16_t* __restrict__ Vt, const float* __restrict__ T,
    const unsigned char* __restrict__ idxg, float* __restrict__ out) {
  __shared__ bf16_t kbuf[2][64 * 72];       // [key][d], double-buffered
  __shared__ bf16_t vtbuf[2][64 * 72];      // [d][key], double-buffered
  __shared__ bf16_t qbuf[4][16 * 72];       // per-wave [qrow][d]
  __shared__ bf16_t pbuf[4][16 * 72];       // per-wave [qrow][key]
  __shared__ float  tbuf[4][16 * 12];       // per-wave [qrow][emb]
  __shared__ unsigned int ibuf[4][256];     // per-wave 16x64 u8 indices

  const int tid  = threadIdx.x;
  const int lane = tid & 31;
  const int wv   = tid >> 5;
  const int half = lane >> 4;
  const int n    = lane & 15;

  const int h  = blockIdx.y;
  const int b  = blockIdx.z;
  const int q0 = blockIdx.x * 64 + wv * 16;

  const size_t headBase = ((size_t)(b * H_ + h)) * S_ * DK_;  // Q,K and Vt
  const bf16_t* Kbase  = K + headBase;
  const bf16_t* VtBase = Vt + headBase;

  // stage this wave's Q tile [16 x 64]
#pragma unroll
  for (int i = 0; i < 4; ++i) {
    int c = i * 32 + lane;                  // 0..127
    int r = c >> 3;
    int cc = (c & 7) * 8;
    *(v8bf*)&qbuf[wv][r * 72 + cc] =
        *(const v8bf*)(Q + headBase + (size_t)(q0 + r) * DK_ + cc);
  }
  // stage this wave's bias table rows [16 x 10]
  for (int i = lane; i < 16 * NE_; i += 32) {
    int r = i / NE_, e = i % NE_;
    tbuf[wv][r * 12 + e] =
        T[((size_t)(b * H_ + h) * S_ + q0 + r) * NE_ + e];
  }

  // issue async K/V tile loads for kt=0 into buffer 0
  {
#pragma unroll
    for (int i = 0; i < 4; ++i) {
      int c = i * 128 + tid;                // 0..511
      int r = c >> 3;
      int cc = (c & 7) * 8;
      async_copy_b128(&kbuf[0][r * 72 + cc], Kbase,
                      (unsigned)(((0 * 64 + r) * DK_ + cc) * sizeof(bf16_t)));
      async_copy_b128(&vtbuf[0][r * 72 + cc], VtBase,
                      (unsigned)((r * S_ + 0 * 64 + cc) * sizeof(bf16_t)));
    }
  }
  __syncthreads();   // qbuf/tbuf visible

  const bf16_t* qb = &qbuf[wv][n * 72];
  v16bf aq0 = make_frag(qb + half * 16,      qb + half * 16 + 8);
  v16bf aq1 = make_frag(qb + 32 + half * 16, qb + 40 + half * 16);

  float mrun[8], lrun[8];
  v8f oacc[4];
#pragma unroll
  for (int r = 0; r < 8; ++r) { mrun[r] = -3.0e38f; lrun[r] = 0.0f; }
#pragma unroll
  for (int g = 0; g < 4; ++g)
#pragma unroll
    for (int r = 0; r < 8; ++r) oacc[g][r] = 0.0f;

  const unsigned char* ib = (const unsigned char*)ibuf[wv];

  for (int kt = 0; kt < S_ / 64; ++kt) {
    const int cur = kt & 1;

    // per-wave idx tile [16 x 64] bytes, as 256 dwords (normal loads)
    {
      const unsigned int* ig = (const unsigned int*)(idxg +
          ((size_t)b * S_ + q0) * S_ + (size_t)kt * 64);
#pragma unroll
      for (int i = 0; i < 8; ++i) {
        int j = i * 32 + lane;              // 0..255
        int r = j >> 4;
        int c4 = j & 15;
        ibuf[wv][r * 16 + c4] = ig[(size_t)r * (S_ / 4) + c4];
      }
    }

    // wait for this wave's async copies of tile kt, then sync all waves
    wait_asynccnt0();
    __syncthreads();

    // issue async copies for tile kt+1 into the other buffer
    if (kt + 1 < S_ / 64) {
      const int nxt = cur ^ 1;
      const int kr = (kt + 1) * 64;
#pragma unroll
      for (int i = 0; i < 4; ++i) {
        int c = i * 128 + tid;
        int r = c >> 3;
        int cc = (c & 7) * 8;
        async_copy_b128(&kbuf[nxt][r * 72 + cc], Kbase,
                        (unsigned)(((kr + r) * DK_ + cc) * sizeof(bf16_t)));
        async_copy_b128(&vtbuf[nxt][r * 72 + cc], VtBase,
                        (unsigned)((r * S_ + kr + cc) * sizeof(bf16_t)));
      }
    }

    // ---- S = Q K^T (4 fragments of 16 key columns) + distance bias ----
    float sfr[4][8];
#pragma unroll
    for (int f = 0; f < 4; ++f) {
      const bf16_t* kb = &kbuf[cur][(f * 16 + n) * 72 + half * 16];
      v8f c;
#pragma unroll
      for (int r = 0; r < 8; ++r) c[r] = 0.0f;
      c = wmma_bf16(aq0, make_frag(kb, kb + 8), c);
      c = wmma_bf16(aq1, make_frag(kb + 32, kb + 40), c);
#pragma unroll
      for (int r = 0; r < 8; ++r) {
        int m = r + 8 * half;
        int id = ib[m * 64 + f * 16 + n];
        sfr[f][r] = (c[r] + tbuf[wv][m * 12 + id]) * 0.125f;
      }
    }

    // ---- online softmax ----
    float mnewr[8], fac[8], psum[8];
#pragma unroll
    for (int r = 0; r < 8; ++r) {
      float mm = fmaxf(fmaxf(sfr[0][r], sfr[1][r]),
                       fmaxf(sfr[2][r], sfr[3][r]));
#pragma unroll
      for (int off = 1; off <= 8; off <<= 1)
        mm = fmaxf(mm, __shfl_xor(mm, off));
      mnewr[r] = fmaxf(mrun[r], mm);
      fac[r] = __expf(mrun[r] - mnewr[r]);
      mrun[r] = mnewr[r];
      lrun[r] *= fac[r];
      psum[r] = 0.0f;
    }
#pragma unroll
    for (int f = 0; f < 4; ++f)
#pragma unroll
      for (int r = 0; r < 8; ++r) {
        float p = __expf(sfr[f][r] - mnewr[r]);
        psum[r] += p;
        pbuf[wv][(r + 8 * half) * 72 + f * 16 + n] = (bf16_t)p;
      }
#pragma unroll
    for (int r = 0; r < 8; ++r) {
#pragma unroll
      for (int off = 1; off <= 8; off <<= 1)
        psum[r] += __shfl_xor(psum[r], off);
      lrun[r] += psum[r];
    }
#pragma unroll
    for (int g = 0; g < 4; ++g)
#pragma unroll
      for (int r = 0; r < 8; ++r) oacc[g][r] *= fac[r];
    __syncthreads();   // pbuf visible to all lanes of this wave

    // ---- O += P V  (K-dim = 64 keys in two chunks of 32) ----
#pragma unroll
    for (int kk2 = 0; kk2 < 2; ++kk2) {
      const bf16_t* pb = &pbuf[wv][n * 72 + kk2 * 32];
      v16bf ap = make_frag(pb + half * 16, pb + half * 16 + 8);
#pragma unroll
      for (int g = 0; g < 4; ++g) {
        const bf16_t* vb = &vtbuf[cur][(g * 16 + n) * 72 + kk2 * 32 + half * 16];
        oacc[g] = wmma_bf16(ap, make_frag(vb, vb + 8), oacc[g]);
      }
    }
  }

  // ---- epilogue: out[b, q0+m, h*64 + g*16 + n] = oacc / l ----
  float* op = out + (size_t)b * S_ * D_;
#pragma unroll
  for (int g = 0; g < 4; ++g)
#pragma unroll
    for (int r = 0; r < 8; ++r) {
      int m = r + 8 * half;
      op[(size_t)(q0 + m) * D_ + h * DK_ + g * 16 + n] =
          oacc[g][r] / lrun[r];
    }
}

// ---------------------------------------------------------------------------
// Launch
// ---------------------------------------------------------------------------
extern "C" void kernel_launch(void* const* d_in, const int* in_sizes, int n_in,
                              void* d_out, int out_size, void* d_ws,
                              size_t ws_size, hipStream_t stream) {
  const float* q    = (const float*)d_in[0];
  const float* k    = (const float*)d_in[1];
  const float* v    = (const float*)d_in[2];
  const float* pos  = (const float*)d_in[3];
  const float* Wq   = (const float*)d_in[4];
  const float* bq   = (const float*)d_in[5];
  const float* Wk   = (const float*)d_in[6];
  const float* bk   = (const float*)d_in[7];
  const float* Wv   = (const float*)d_in[8];
  const float* bv   = (const float*)d_in[9];
  const float* emb  = (const float*)d_in[10];
  float* out = (float*)d_out;

  const size_t nQ = (size_t)B_ * H_ * S_ * DK_;      // 3,145,728
  bf16_t* Qb  = (bf16_t*)d_ws;
  bf16_t* Kb  = Qb + nQ;
  bf16_t* Vtb = Kb + nQ;
  float*  T   = (float*)(Vtb + nQ);                  // B*H*S*10 floats
  unsigned char* idx = (unsigned char*)(T + (size_t)B_ * H_ * S_ * NE_);

  // 1) projections (z: 0=Q, 1=K, 2=V-transposed)
  dim3 gp((B_ * S_) / 64, D_ / 64, 3);
  proj_kernel<<<gp, 128, 0, stream>>>(q, k, v, Wq, Wk, Wv, bq, bk, bv,
                                      Qb, Kb, Vtb);
  // 2) bias table
  int nT = B_ * H_ * S_ * NE_;
  bias_table_kernel<<<(nT + 255) / 256, 256, 0, stream>>>(Qb, emb, T);
  // 3) distance -> embedding index
  long nI = (long)B_ * S_ * S_;
  idx_kernel<<<(int)((nI + 255) / 256), 256, 0, stream>>>(pos, idx, nI);
  // 4) attention
  dim3 ga(S_ / 64, H_, B_);
  attn_kernel<<<ga, 128, 0, stream>>>(Qb, Kb, Vtb, T, idx, out);
}
